// Module_2_68341519613953
// MI455X (gfx1250) — matrix operand order repaired and checked by reading, not compiled, not run on hardware
//
#include <hip/hip_runtime.h>
#include <hip/hip_bf16.h>

// Attention: x[8,2048,512] f32; Wq/Wk/Wv/Wo [512,512]; biases [512].
// All matmuls via V_WMMA_F32_16X16X32_F16 (f16 in, f32 accumulate).
// V tiles in the P@V stage are staged LDS-side with global_load_async_to_lds.

#define EMBED 512
#define SEQ   2048
#define BATCH 8
#define MTOT  (BATCH * SEQ)   // 16384

typedef __attribute__((ext_vector_type(16))) _Float16 v16h;
typedef __attribute__((ext_vector_type(8)))  float    v8f;

#define WMMA_F16(a, b, c) \
    __builtin_amdgcn_wmma_f32_16x16x32_f16(false, (a), false, (b), (short)0, (c), false, false)

// ---- fragment loaders (contiguous k order: k = 16*(lane/16) + element) -------------

// A fragment 16x32 f16, row-major source, leading dim ld.
// lanes 0-15: M=lane, K=0..15
// lanes16-31: M=lane-16, K=16..31
__device__ __forceinline__ v16h load_a_f16(const _Float16* A, int ld, int row0,
                                           int k0, int lane) {
    int m  = lane & 15;
    int kb = (lane >> 4) << 4;            // 0 or 16
    const _Float16* p = A + (size_t)(row0 + m) * ld + k0 + kb;
    v16h a;
#pragma unroll
    for (int i = 0; i < 16; ++i) a[i] = p[i];
    return a;
}

// B fragment 32x16: logical B[k][n] = src[(n0+n)*ld + k0+k]  (source holds the
// n index as rows; contraction axis k is contiguous -> two b128 loads/lane).
// lanes 0-15: N=lane, K=0..15; lanes 16-31: N=lane-16, K=16..31
__device__ __forceinline__ v16h load_bT_f16(const _Float16* B, int ld, int k0,
                                            int n0, int lane) {
    int n  = lane & 15;
    int kb = (lane >> 4) << 4;
    const _Float16* p = B + (size_t)(n0 + n) * ld + k0 + kb;
    v16h b;
#pragma unroll
    for (int i = 0; i < 16; ++i) b[i] = p[i];
    return b;
}

// ---- elementwise converts ---------------------------------------------------

__global__ void convert_f32_to_f16(const float* __restrict__ src,
                                   _Float16* __restrict__ dst, int n) {
    int i = blockIdx.x * blockDim.x + threadIdx.x;
    if (i < n) dst[i] = (_Float16)src[i];
}

// W[k][n] f32 -> Wt[n][k] f16   (both EMBED x EMBED)
__global__ void convert_transpose_w(const float* __restrict__ src,
                                    _Float16* __restrict__ dst) {
    int i = blockIdx.x * blockDim.x + threadIdx.x;   // over EMBED*EMBED
    int n = i >> 9, k = i & (EMBED - 1);
    dst[i] = (_Float16)src[(size_t)k * EMBED + n];
}

// ---- generic GEMM: C = A(f16) @ Wt(f16, pre-transposed [N][K]) + bias -------
// Each wave computes a 32x64 tile; block = 4 waves covering 32(M) x 256(N).
// MODE 0: f16 row-major out; MODE 1: f32 row-major out;
// MODE 2: f16 out transposed per batch -> Ch[b][col][m], b = row/SEQ.

template <int MODE>
__global__ __launch_bounds__(128)
void gemm_f16(const _Float16* __restrict__ A, const _Float16* __restrict__ Wt,
              const float* __restrict__ bias, float* __restrict__ Cf,
              _Float16* __restrict__ Ch, int M, int N, int K) {
    int lane = threadIdx.x & 31;
    int wave = threadIdx.x >> 5;
    int m0 = blockIdx.x * 32;
    int n0 = blockIdx.y * 256 + wave * 64;
    if (m0 >= M || n0 >= N) return;

    v8f acc[2][4] = {};
    for (int k0 = 0; k0 < K; k0 += 32) {
        if (k0 + 32 < K)  // hint next A panel into cache (global_prefetch_b8)
            __builtin_prefetch(A + (size_t)(m0 + lane) * K + k0 + 32, 0, 0);
        v16h a0 = load_a_f16(A, K, m0,      k0, lane);
        v16h a1 = load_a_f16(A, K, m0 + 16, k0, lane);
#pragma unroll
        for (int t = 0; t < 4; ++t) {
            v16h bf = load_bT_f16(Wt, K, k0, n0 + t * 16, lane);
            acc[0][t] = WMMA_F16(a0, bf, acc[0][t]);
            acc[1][t] = WMMA_F16(a1, bf, acc[1][t]);
        }
    }

    int n  = lane & 15;
    int mb = (lane >> 4) << 3;            // 0 or 8
#pragma unroll
    for (int mi = 0; mi < 2; ++mi)
#pragma unroll
        for (int t = 0; t < 4; ++t)
#pragma unroll
            for (int r = 0; r < 8; ++r) {
                int row = m0 + mi * 16 + mb + r;
                int col = n0 + t * 16 + n;
                float v = acc[mi][t][r] + bias[col];
                if constexpr (MODE == 1) {
                    Cf[(size_t)row * N + col] = v;
                } else if constexpr (MODE == 0) {
                    Ch[(size_t)row * N + col] = (_Float16)v;
                } else {               // transposed per-batch [EMBED][SEQ]
                    int bb = row >> 11;                // row / SEQ
                    int m  = row & (SEQ - 1);
                    Ch[((size_t)bb * N + col) * SEQ + m] = (_Float16)v;
                }
            }
}

// ---- fused attention core: one workgroup per (batch, 16-query block) --------
// LDS layout (dynamic, ~257KB of the 320KB WGP LDS):
//   [0,128K)       S[16][2048] f32 scores
//   [128K,192K)    P[16][2048] f16 softmax probabilities (compact)
//   [192K,224K)    Vt tile buffer 0  ([512 d][32 m] f16)
//   [224K,256K)    Vt tile buffer 1
//   [256K,...)     reduction scratch
#define SMEM_S     0
#define SMEM_P     (16 * SEQ * 4)
#define SMEM_V0    (SMEM_P + 16 * SEQ * 2)
#define SMEM_V1    (SMEM_V0 + 32 * EMBED * 2)
#define SMEM_RED   (SMEM_V1 + 32 * EMBED * 2)
#define SMEM_TOTAL (SMEM_RED + (256 + 32) * 4)

// B fragment from LDS-resident Vt tile [EMBED rows][32 halves], n = d index.
__device__ __forceinline__ v16h load_b_ldsVt(const _Float16* Vs, int n0, int lane) {
    int n  = lane & 15;
    int kb = (lane >> 4) << 4;
    const _Float16* p = Vs + (size_t)(n0 + n) * 32 + kb;
    v16h b;
#pragma unroll
    for (int i = 0; i < 16; ++i) b[i] = p[i];
    return b;
}

// Stage a 32-key slice of transposed V (vt[b][d][m]) into LDS as [512][32]
// halves using the CDNA5 async global->LDS DMA path (tracked by ASYNCcnt).
__device__ __forceinline__ void stage_vt_tile(const _Float16* Vtb, int k0,
                                              char* lds_dst, int tid) {
    unsigned l = (unsigned)(size_t)lds_dst;
#pragma unroll
    for (int it = 0; it < 8; ++it) {
        int c   = it * 256 + tid;      // 0..2047 16B chunks
        int row = c >> 2;              // d index 0..511
        int sub = c & 3;               // 16B chunk within the 64B row
        const char* g = (const char*)(Vtb + (size_t)row * SEQ + k0) + sub * 16;
        unsigned    d = l + (unsigned)(row * 64 + sub * 16);
        asm volatile("global_load_async_to_lds_b128 %0, %1, off"
                     :: "v"(d), "v"(g) : "memory");
    }
}

__device__ __forceinline__ void wait_async_all() {
    asm volatile("s_wait_asynccnt 0x0" ::: "memory");
}

__global__ __launch_bounds__(256)
void attn_core(const _Float16* __restrict__ qh, const _Float16* __restrict__ kh,
               const _Float16* __restrict__ vt, _Float16* __restrict__ oh) {
    extern __shared__ char smem[];
    float*     Sf     = (float*)(smem + SMEM_S);        // 16*SEQ f32
    _Float16*  Pp     = (_Float16*)(smem + SMEM_P);     // 16*SEQ f16
    char*      Vb0    = smem + SMEM_V0;
    char*      Vb1    = smem + SMEM_V1;
    float*     red    = (float*)(smem + SMEM_RED);      // 16*16 partials
    float*     rowmax = red + 256;                      // 16
    float*     rowinv = rowmax + 16;                    // 16

    const int tid  = threadIdx.x;
    const int lane = tid & 31;
    const int wave = tid >> 5;                          // 0..7
    const int b    = blockIdx.y;
    const int q0   = blockIdx.x * 16;
    const float scale = 0.044194173824159216f;          // 512^-0.5

    const _Float16* Qb  = qh + ((size_t)b * SEQ + q0) * EMBED;
    const _Float16* Kb  = kh + (size_t)b * SEQ * EMBED;
    const _Float16* Vtb = vt + (size_t)b * EMBED * SEQ;

    // ---- phase 1: S = Q @ K^T, staged into LDS ----
    v16h aq[16];                                        // Q block: 16 x 512
#pragma unroll
    for (int kk = 0; kk < 16; ++kk)
        aq[kk] = load_a_f16(Qb, EMBED, 0, kk * 32, lane);

    for (int t = 0; t < 16; ++t) {                      // 16 key tiles per wave
        int n0 = wave * 256 + t * 16;
        v8f c = {};
#pragma unroll
        for (int kk = 0; kk < 16; ++kk)
            c = WMMA_F16(aq[kk], load_bT_f16(Kb, EMBED, kk * 32, n0, lane), c);
        int n  = lane & 15;
        int mb = (lane >> 4) << 3;
#pragma unroll
        for (int r = 0; r < 8; ++r)
            Sf[(size_t)(mb + r) * SEQ + n0 + n] = c[r];
    }
    __syncthreads();

    // ---- phase 2: softmax over each of the 16 rows (16 threads per row) ----
    {
        int r = tid >> 4, sub = tid & 15;
        const float* row = Sf + (size_t)r * SEQ;
        float mx = -__builtin_inff();
        for (int c = sub * 128; c < sub * 128 + 128; ++c)
            mx = fmaxf(mx, row[c]);
        red[r * 16 + sub] = mx;
        __syncthreads();
        if (sub == 0) {
            float m = red[r * 16];
#pragma unroll
            for (int i = 1; i < 16; ++i) m = fmaxf(m, red[r * 16 + i]);
            rowmax[r] = m * scale;
        }
        __syncthreads();
        float M = rowmax[r];
        float sm = 0.f;
        for (int c = sub * 128; c < sub * 128 + 128; ++c)
            sm += __expf(scale * row[c] - M);
        red[r * 16 + sub] = sm;
        __syncthreads();
        if (sub == 0) {
            float s = 0.f;
#pragma unroll
            for (int i = 0; i < 16; ++i) s += red[r * 16 + i];
            rowinv[r] = 1.0f / s;
        }
        __syncthreads();
    }

    // normalize into the compact f16 P buffer
    for (int idx = tid; idx < 16 * SEQ; idx += 256) {
        int rr = idx >> 11;                             // / SEQ
        Pp[idx] = (_Float16)(__expf(scale * Sf[idx] - rowmax[rr]) * rowinv[rr]);
    }
    __syncthreads();

    // ---- phase 3: O = P @ V, Vt tiles double-buffered via async LDS DMA ----
    int n0 = wave * 64;
    v8f acc[4] = {};

    stage_vt_tile(Vtb, 0, Vb0, tid);
    wait_async_all();
    __syncthreads();

    for (int i = 0; i < SEQ / 32; ++i) {
        int k0 = i * 32;
        const _Float16* curV = (const _Float16*)((i & 1) ? Vb1 : Vb0);
        char*           nxtV = (i & 1) ? Vb0 : Vb1;
        if (i + 1 < SEQ / 32)
            stage_vt_tile(Vtb, k0 + 32, nxtV, tid);     // overlap DMA w/ WMMA

        v16h ap = load_a_f16(Pp, SEQ, 0, k0, lane);     // ds_load_b128 x2
#pragma unroll
        for (int t = 0; t < 4; ++t)
            acc[t] = WMMA_F16(ap, load_b_ldsVt(curV, n0 + t * 16, lane), acc[t]);

        wait_async_all();
        __syncthreads();
    }

    _Float16* Ob = oh + ((size_t)b * SEQ + q0) * EMBED;
    int n  = lane & 15;
    int mb = (lane >> 4) << 3;
#pragma unroll
    for (int t = 0; t < 4; ++t)
#pragma unroll
        for (int r = 0; r < 8; ++r)
            Ob[(size_t)(mb + r) * EMBED + n0 + t * 16 + n] = (_Float16)acc[t][r];
}

// ---- host side --------------------------------------------------------------

extern "C" void kernel_launch(void* const* d_in, const int* in_sizes, int n_in,
                              void* d_out, int out_size, void* d_ws, size_t ws_size,
                              hipStream_t stream) {
    const float* x  = (const float*)d_in[0];
    const float* Wq = (const float*)d_in[1];
    const float* bq = (const float*)d_in[2];
    const float* Wk = (const float*)d_in[3];
    const float* bk = (const float*)d_in[4];
    const float* Wv = (const float*)d_in[5];
    const float* bv = (const float*)d_in[6];
    const float* Wo = (const float*)d_in[7];
    const float* bo = (const float*)d_in[8];

    const size_t elems  = (size_t)MTOT * EMBED;         // 8.4M
    const size_t welems = (size_t)EMBED * EMBED;        // 262144
    _Float16* xh  = (_Float16*)d_ws;                    // 16 MB each
    _Float16* qh  = xh + elems;
    _Float16* kh  = qh + elems;
    _Float16* vt  = kh + elems;                         // transposed V
    _Float16* oh  = vt + elems;
    _Float16* wqt = oh + elems;                         // 0.5 MB each
    _Float16* wkt = wqt + welems;
    _Float16* wvt = wkt + welems;
    _Float16* wot = wvt + welems;

    convert_f32_to_f16<<<(int)((elems + 255) / 256), 256, 0, stream>>>(
        x, xh, (int)elems);
    int wgrid = (int)((welems + 255) / 256);
    convert_transpose_w<<<wgrid, 256, 0, stream>>>(Wq, wqt);
    convert_transpose_w<<<wgrid, 256, 0, stream>>>(Wk, wkt);
    convert_transpose_w<<<wgrid, 256, 0, stream>>>(Wv, wvt);
    convert_transpose_w<<<wgrid, 256, 0, stream>>>(Wo, wot);

    dim3 gg(MTOT / 32, EMBED / 256);                    // 512 x 2
    gemm_f16<0><<<gg, 128, 0, stream>>>(xh, wqt, bq, nullptr, qh,
                                        MTOT, EMBED, EMBED);
    gemm_f16<0><<<gg, 128, 0, stream>>>(xh, wkt, bk, nullptr, kh,
                                        MTOT, EMBED, EMBED);
    gemm_f16<2><<<gg, 128, 0, stream>>>(xh, wvt, bv, nullptr, vt,
                                        MTOT, EMBED, EMBED);

    attn_core<<<dim3(SEQ / 16, BATCH), 256, SMEM_TOTAL, stream>>>(qh, kh, vt, oh);

    gemm_f16<1><<<gg, 128, 0, stream>>>(oh, wot, bo, (float*)d_out, nullptr,
                                        MTOT, EMBED, EMBED);
}
